// LabelAttention_88699664597684
// MI455X (gfx1250) — matrix operand order repaired and checked by reading, not compiled, not run on hardware
//
#include <hip/hip_runtime.h>

#define BB 2
#define SS 1024
#define DD 768
#define HH 4
#define LL 8921
#define LBLK 8          // labels per block (x4 heads = 32 WMMA columns)
#define NC 32           // columns per block (two 16-wide WMMA groups)
#define UST 776         // padded LDS stride for U tile (bank-conflict-free)
#define AST 1032        // padded LDS stride for score/alpha columns

typedef __attribute__((ext_vector_type(16))) __bf16 v16bf;
typedef __attribute__((ext_vector_type(8)))  float  v8f;
typedef __attribute__((ext_vector_type(4)))  unsigned int u32x4;

union FragBF {
  v16bf v;
  u32x4 q[2];
};

__device__ __forceinline__ unsigned short f2bf(float f) {
  unsigned int u = __float_as_uint(f);
  u += 0x7FFFu + ((u >> 16) & 1u);          // round-to-nearest-even
  return (unsigned short)(u >> 16);
}
__device__ __forceinline__ float bf2f(unsigned short h) {
  return __uint_as_float(((unsigned int)h) << 16);
}

// ---------------------------------------------------------------------------
// Kernel 0a: W_w f32 -> bf16
__global__ void k_conv_w(const float* __restrict__ W, unsigned short* __restrict__ Wb) {
  int i = blockIdx.x * blockDim.x + threadIdx.x;
  if (i < DD * DD) Wb[i] = f2bf(W[i]);
}

// Kernel 0b: h f32 -> bf16 row-major [B][S][D] and transposed [B][D][S]
__global__ void k_conv_h(const float* __restrict__ h,
                         unsigned short* __restrict__ Hb,
                         unsigned short* __restrict__ HT) {
  int i = blockIdx.x * blockDim.x + threadIdx.x;
  if (i >= BB * SS * DD) return;
  unsigned short v = f2bf(h[i]);
  Hb[i] = v;
  int d = i % DD;
  int s = (i / DD) % SS;
  int b = i / (DD * SS);
  HT[((size_t)b * DD + d) * SS + s] = v;
}

// ---------------------------------------------------------------------------
// Kernel 1: Z = tanh(H @ W^T + bias), bf16 in / bf16 out, f32 accumulate.
__global__ void __launch_bounds__(128) k_zgemm(const unsigned short* __restrict__ Hb,
                                               const unsigned short* __restrict__ Wb,
                                               const float* __restrict__ Wbias,
                                               unsigned short* __restrict__ Zb) {
  const int lane  = threadIdx.x & 31;
  const int wave  = threadIdx.x >> 5;
  const int l15   = lane & 15;
  const int hi    = lane >> 4;                 // 0 or 1 (half-wave)
  const int mbase = blockIdx.x * 16;           // output row tile (B*S rows)
  const int ebase = (blockIdx.y * 4 + wave) * 16;

  v8f acc = {};
  const unsigned short* arow = Hb + (size_t)(mbase + l15) * DD;
  const unsigned short* brow = Wb + (size_t)(ebase + l15) * DD;   // W row e = col of W^T
  for (int k = 0; k < DD; k += 32) {
    FragBF a, b;
    a.q[0] = *(const u32x4*)(arow + k + hi * 8);
    a.q[1] = *(const u32x4*)(arow + k + hi * 8 + 16);
    b.q[0] = *(const u32x4*)(brow + k + hi * 8);
    b.q[1] = *(const u32x4*)(brow + k + hi * 8 + 16);
    acc = __builtin_amdgcn_wmma_f32_16x16x32_bf16(false, a.v, false, b.v,
                                                  (short)0, acc, false, false);
  }
  const int e = ebase + l15;
  const float bias = Wbias[e];
#pragma unroll
  for (int j = 0; j < 8; ++j) {
    int m = mbase + j + hi * 8;                // C layout: VGPR j -> row j (+8 hi half)
    Zb[(size_t)m * DD + e] = f2bf(tanhf(acc[j] + bias));
  }
}

// ---------------------------------------------------------------------------
// Kernel 2: fused label attention for one (batch, 8-label block).
// Each streamed Z / HT fragment now feeds TWO 16-column WMMA groups, halving
// L2 traffic on the bandwidth-critical operand (32 flops/byte).
__global__ void __launch_bounds__(128) k_label_attn(
    const unsigned short* __restrict__ Zb,    // [B][S][D] bf16
    const unsigned short* __restrict__ HT,    // [B][D][S] bf16
    const float* __restrict__ label_feat,     // [L*H][D]
    const int*  __restrict__ amask,           // [B][S]
    const float* __restrict__ C_w,            // [L][D]
    const float* __restrict__ C_b,            // [L]
    float* __restrict__ out)                  // [B][L]
{
  __shared__ __align__(16) unsigned short sU[NC * UST];   // label queries, bf16 (49.7 KB)
  __shared__ __align__(16) unsigned short sA[NC * AST];   // scores -> alpha, bf16 (66 KB)
  __shared__ signed char sMask[SS];
  __shared__ float sRed[NC * 4];
  __shared__ float sMax[NC];
  __shared__ float sSum[NC];
  __shared__ float sLogit[LBLK];

  const int tid   = threadIdx.x;
  const int lane  = tid & 31;
  const int wave  = tid >> 5;
  const int l15   = lane & 15;
  const int hi    = lane >> 4;
  const int lBase = blockIdx.x * LBLK;
  const int b     = blockIdx.y;

  // ---- stage U tile (32 (l,h) columns x 768), mask bits, init ----
  for (int i = tid; i < NC * DD; i += 128) {
    int col = i / DD, d = i % DD;
    int row = lBase * HH + col;               // (lBase + col/4)*H + col%4
    float f = (row < LL * HH) ? label_feat[(size_t)row * DD + d] : 0.0f;
    sU[col * UST + d] = f2bf(f);
  }
  for (int i = tid; i < SS; i += 128) sMask[i] = (signed char)(amask[b * SS + i] != 0);
  if (tid < LBLK) sLogit[tid] = 0.0f;
  __syncthreads();

  // ---- phase 1: score[s, col] = Z[s,:] . U[col,:]  (A reused for 2 B tiles) ----
  const unsigned short* zBase  = Zb + (size_t)b * SS * DD;
  const unsigned short* bcolU0 = sU + l15 * UST;
  const unsigned short* bcolU1 = sU + (l15 + 16) * UST;
  for (int t = 0; t < 16; ++t) {
    const int mrow = wave * 256 + t * 16;
    const unsigned short* arow = zBase + (size_t)(mrow + l15) * DD;
    if (t + 1 < 16)
      __builtin_prefetch(zBase + (size_t)(mrow + 16 + l15) * DD, 0, 2);
    v8f acc0 = {}, acc1 = {};
    for (int k = 0; k < DD; k += 32) {
      FragBF a, b0, b1;
      a.q[0]  = *(const u32x4*)(arow + k + hi * 8);
      a.q[1]  = *(const u32x4*)(arow + k + hi * 8 + 16);
      b0.q[0] = *(const u32x4*)(bcolU0 + k + hi * 8);
      b0.q[1] = *(const u32x4*)(bcolU0 + k + hi * 8 + 16);
      b1.q[0] = *(const u32x4*)(bcolU1 + k + hi * 8);
      b1.q[1] = *(const u32x4*)(bcolU1 + k + hi * 8 + 16);
      acc0 = __builtin_amdgcn_wmma_f32_16x16x32_bf16(false, a.v, false, b0.v,
                                                     (short)0, acc0, false, false);
      acc1 = __builtin_amdgcn_wmma_f32_16x16x32_bf16(false, a.v, false, b1.v,
                                                     (short)0, acc1, false, false);
    }
#pragma unroll
    for (int j = 0; j < 8; ++j) {
      int s = mrow + j + hi * 8;
      float m0 = sMask[s] ? acc0[j] : -1.0e6f;
      float m1 = sMask[s] ? acc1[j] : -1.0e6f;
      sA[l15 * AST + s]        = f2bf(m0);    // column-major, padded stride
      sA[(l15 + 16) * AST + s] = f2bf(m1);
    }
  }
  __syncthreads();

  // ---- phase 2: per-column softmax over S, normalized alpha written in place ----
  {
    const int col = tid & 31;
    const int chunk = tid >> 5;               // 4 chunks of 256 seq positions
    unsigned short* cp = sA + col * AST + chunk * 256;
    float mx = -3.0e38f;
    for (int i = 0; i < 256; ++i) mx = fmaxf(mx, bf2f(cp[i]));
    sRed[col * 4 + chunk] = mx;
    __syncthreads();
    if (tid < NC) {
      float m2 = sRed[tid * 4];
      for (int i = 1; i < 4; ++i) m2 = fmaxf(m2, sRed[tid * 4 + i]);
      sMax[tid] = m2;
    }
    __syncthreads();
    const float cmax = sMax[col];
    float sum = 0.0f;
    for (int i = 0; i < 256; ++i) sum += __expf(bf2f(cp[i]) - cmax);
    sRed[col * 4 + chunk] = sum;
    __syncthreads();
    if (tid < NC) {
      float s2 = 0.0f;
      for (int i = 0; i < 4; ++i) s2 += sRed[tid * 4 + i];
      sSum[tid] = s2;
    }
    __syncthreads();
    const float rinv = 1.0f / sSum[col];
    for (int i = 0; i < 256; ++i) cp[i] = f2bf(__expf(bf2f(cp[i]) - cmax) * rinv);
  }
  __syncthreads();

  // ---- phase 3: m[d, col] = sum_s HT[d,s]*alpha[s,col]; head-max; dot C_w ----
  const unsigned short* htBase = HT + (size_t)b * DD * SS;
  const unsigned short* bcolA0 = sA + l15 * AST;
  const unsigned short* bcolA1 = sA + (l15 + 16) * AST;
  const int  lab0 = l15 >> 2;                       // label of column group 0
  const int  lab1 = lab0 + 4;                       // label of column group 1
  const int  gl0  = lBase + lab0;
  const int  gl1  = lBase + lab1;
  const bool wlane = ((lane & 3) == 0);
  const bool w0 = wlane && (gl0 < LL);
  const bool w1 = wlane && (gl1 < LL);
  float partial0 = 0.0f, partial1 = 0.0f;
  for (int t = 0; t < 12; ++t) {
    const int dbase = wave * 192 + t * 16;
    const unsigned short* arow = htBase + (size_t)(dbase + l15) * SS;
    if (t + 1 < 12)
      __builtin_prefetch(htBase + (size_t)(dbase + 16 + l15) * SS, 0, 2);
    v8f acc0 = {}, acc1 = {};
    for (int k = 0; k < SS; k += 32) {
      FragBF a, b0, b1;
      a.q[0]  = *(const u32x4*)(arow + k + hi * 8);
      a.q[1]  = *(const u32x4*)(arow + k + hi * 8 + 16);
      b0.q[0] = *(const u32x4*)(bcolA0 + k + hi * 8);
      b0.q[1] = *(const u32x4*)(bcolA0 + k + hi * 8 + 16);
      b1.q[0] = *(const u32x4*)(bcolA1 + k + hi * 8);
      b1.q[1] = *(const u32x4*)(bcolA1 + k + hi * 8 + 16);
      acc0 = __builtin_amdgcn_wmma_f32_16x16x32_bf16(false, a.v, false, b0.v,
                                                     (short)0, acc0, false, false);
      acc1 = __builtin_amdgcn_wmma_f32_16x16x32_bf16(false, a.v, false, b1.v,
                                                     (short)0, acc1, false, false);
    }
#pragma unroll
    for (int j = 0; j < 8; ++j) {
      const int d = dbase + j + hi * 8;
      float v0 = acc0[j];
      v0 = fmaxf(v0, __shfl_xor(v0, 1, 32));        // max over 4 head columns
      v0 = fmaxf(v0, __shfl_xor(v0, 2, 32));
      float v1 = acc1[j];
      v1 = fmaxf(v1, __shfl_xor(v1, 1, 32));
      v1 = fmaxf(v1, __shfl_xor(v1, 2, 32));
      if (w0) partial0 += v0 * C_w[(size_t)gl0 * DD + d];
      if (w1) partial1 += v1 * C_w[(size_t)gl1 * DD + d];
    }
  }
  if (w0) atomicAdd(&sLogit[lab0], partial0);
  if (w1) atomicAdd(&sLogit[lab1], partial1);
  __syncthreads();
  if (tid < LBLK && (lBase + tid) < LL)
    out[(size_t)b * LL + lBase + tid] = sLogit[tid] + C_b[lBase + tid];
}

// ---------------------------------------------------------------------------
extern "C" void kernel_launch(void* const* d_in, const int* in_sizes, int n_in,
                              void* d_out, int out_size, void* d_ws, size_t ws_size,
                              hipStream_t stream) {
  const float* h   = (const float*)d_in[0];
  const int*   am  = (const int*)d_in[1];
  const float* lf  = (const float*)d_in[2];
  const float* Ww  = (const float*)d_in[3];
  const float* Wbv = (const float*)d_in[4];
  const float* Cw  = (const float*)d_in[5];
  const float* Cb  = (const float*)d_in[6];
  float* out = (float*)d_out;

  unsigned short* wsp  = (unsigned short*)d_ws;
  unsigned short* Wb16 = wsp;                       //  768*768       bf16
  unsigned short* Hb16 = Wb16 + DD * DD;            //  B*S*D         bf16
  unsigned short* HT   = Hb16 + BB * SS * DD;       //  B*D*S         bf16
  unsigned short* Zb   = HT   + BB * SS * DD;       //  B*S*D         bf16
  // total workspace: ~10.6 MB

  k_conv_w<<<(DD * DD + 255) / 256, 256, 0, stream>>>(Ww, Wb16);
  k_conv_h<<<(BB * SS * DD + 255) / 256, 256, 0, stream>>>(h, Hb16, HT);
  k_zgemm<<<dim3((BB * SS) / 16, (DD / 16) / 4), 128, 0, stream>>>(Hb16, Wb16, Wbv, Zb);
  k_label_attn<<<dim3((LL + LBLK - 1) / LBLK, BB), 128, 0, stream>>>(
      Zb, HT, lf, am, Cw, Cb, out);
}
